// MultiHeadAttention_Enc_Dec_781684048475
// MI455X (gfx1250) — compile-verified, matrix operand order checked by reading, not run on hardware
//
#include <hip/hip_runtime.h>

// ---------------------------------------------------------------------------
// MHA (enc-dec quirk version) for gfx1250 / MI455X.
// All matmuls via v_wmma_f32_16x16x32_bf16 (fp32 accumulate), software
// pipelined with ping-pong fragment buffers (2x unroll -> no register copies).
// ---------------------------------------------------------------------------

typedef __bf16 bf16;
typedef __attribute__((ext_vector_type(16))) __bf16 v16bf;
typedef __attribute__((ext_vector_type(8)))  __bf16 v8bf;
typedef __attribute__((ext_vector_type(8)))  float  v8f;
typedef __attribute__((ext_vector_type(4)))  float  v4f;

#define EMB   1024
#define HEADS 16
#define DH    64          // EMB / HEADS
#define BB    4
#define SS    2048
#define MROWS (BB * SS)   // 8192

#define WMMA_BF16(a, b, c) \
  __builtin_amdgcn_wmma_f32_16x16x32_bf16(false, (a), false, (b), (short)0, (c), false, false)

// Load a 16(row) x 32(k) bf16 fragment in the CDNA5 A/B register layout.
// Lane l (l<16): row = l,   halves 0..7 -> k = 0..7,   halves 8..15 -> k = 16..23
// Lane l (>=16): row = l-16, halves 0..7 -> k = 8..15, halves 8..15 -> k = 24..31
// i.e. per lane: two contiguous 16-byte chunks at k = 8g and k = 16+8g.
__device__ __forceinline__ v16bf load_frag(const bf16* tile, int stride) {
  int lane = threadIdx.x & 31;
  const bf16* p = tile + (size_t)(lane & 15) * stride + 8 * (lane >> 4);
  v8bf lo = *reinterpret_cast<const v8bf*>(p);
  v8bf hi = *reinterpret_cast<const v8bf*>(p + 16);
  return __builtin_shufflevector(lo, hi, 0, 1, 2, 3, 4, 5, 6, 7,
                                 8, 9, 10, 11, 12, 13, 14, 15);
}

// ---------------------------------------------------------------------------
// fp32 -> bf16 conversion, 8 elements / lane (b128 in x2, b128 out)
// ---------------------------------------------------------------------------
__global__ void cvt_f32_to_bf16(const float* __restrict__ in,
                                unsigned short* __restrict__ out, int n) {
  int i = (blockIdx.x * blockDim.x + threadIdx.x) * 8;
  if (i < n) {
    v4f x0 = *reinterpret_cast<const v4f*>(in + i);
    v4f x1 = *reinterpret_cast<const v4f*>(in + i + 4);
    v8bf o;
#pragma unroll
    for (int j = 0; j < 4; ++j) {
      o[j]     = (bf16)x0[j];
      o[j + 4] = (bf16)x1[j];
    }
    *reinterpret_cast<v8bf*>(out + i) = o;
  }
}

// ---------------------------------------------------------------------------
// Y[M=8192, N=1024] = A[M,K=1024] @ W[N,K]^T + bias[N]
// mode 0: store bf16 at q[((b*H+h)*S + s)*DH + d]          (Q, head-split)
// mode 1: store bf16 at k[((b*H+h)*S + s)*DH + d]          (K, head-split)
// mode 2: store bf16 at vt[((b*H+h)*DH + d)*S + s]         (V, transposed)
// mode 3: store fp32 at out[m*EMB + n]                     (final projection)
// Block = 128 threads (4 waves); each wave computes a 32x64 output tile
// (2 A-fragments x 4 B-fragments -> 8 WMMAs per 32-k step). K-loop is
// unrolled 2x with ping-pong fragment sets: every fragment register is
// written by a global load, never a copy.
// ---------------------------------------------------------------------------
__global__ __launch_bounds__(128) void gemm_wmma(
    const unsigned short* __restrict__ Au, const unsigned short* __restrict__ Wu,
    const float* __restrict__ bias, unsigned short* __restrict__ outBu,
    float* __restrict__ outF, int mode) {
  const bf16* A = reinterpret_cast<const bf16*>(Au);
  const bf16* W = reinterpret_cast<const bf16*>(Wu);
  bf16* outB = reinterpret_cast<bf16*>(outBu);

  int wave = threadIdx.x >> 5;
  int lane = threadIdx.x & 31;
  int rBase = blockIdx.x * 128 + wave * 32;  // output rows (32 per wave)
  int cBase = blockIdx.y * 64;               // output cols (64 per wave)

  v8f acc[2][4];
#pragma unroll
  for (int rg = 0; rg < 2; ++rg)
#pragma unroll
    for (int n = 0; n < 4; ++n)
#pragma unroll
      for (int r = 0; r < 8; ++r) acc[rg][n][r] = 0.0f;

  const bf16* Ar0 = A + (size_t)rBase * EMB;
  const bf16* Ar1 = A + (size_t)(rBase + 16) * EMB;
  const bf16* Wc = W + (size_t)cBase * EMB;

  // set X: even k-steps, set Y: odd k-steps (ping-pong, no copies)
  v16bf a0x = load_frag(Ar0, EMB);
  v16bf a1x = load_frag(Ar1, EMB);
  v16bf bx[4];
#pragma unroll
  for (int n = 0; n < 4; ++n) bx[n] = load_frag(Wc + (size_t)(n * 16) * EMB, EMB);

  for (int k0 = 0; k0 < EMB; k0 += 64) {
    // prefetch odd step (k0+32) -- always in range (EMB % 64 == 0)
    v16bf a0y = load_frag(Ar0 + k0 + 32, EMB);
    v16bf a1y = load_frag(Ar1 + k0 + 32, EMB);
    v16bf by[4];
#pragma unroll
    for (int n = 0; n < 4; ++n)
      by[n] = load_frag(Wc + (size_t)(n * 16) * EMB + k0 + 32, EMB);

    // consume even step
#pragma unroll
    for (int n = 0; n < 4; ++n) {
      acc[0][n] = WMMA_BF16(a0x, bx[n], acc[0][n]);
      acc[1][n] = WMMA_BF16(a1x, bx[n], acc[1][n]);
    }

    // prefetch next even step (k0+64), uniform branch
    if (k0 + 64 < EMB) {
      a0x = load_frag(Ar0 + k0 + 64, EMB);
      a1x = load_frag(Ar1 + k0 + 64, EMB);
#pragma unroll
      for (int n = 0; n < 4; ++n)
        bx[n] = load_frag(Wc + (size_t)(n * 16) * EMB + k0 + 64, EMB);
    }

    // consume odd step
#pragma unroll
    for (int n = 0; n < 4; ++n) {
      acc[0][n] = WMMA_BF16(a0y, by[n], acc[0][n]);
      acc[1][n] = WMMA_BF16(a1y, by[n], acc[1][n]);
    }
  }

  int g = lane >> 4;
  int c = lane & 15;
#pragma unroll
  for (int rg = 0; rg < 2; ++rg)
#pragma unroll
    for (int n = 0; n < 4; ++n) {
      int nc = cBase + n * 16 + c;
      float bvv = bias[nc];
#pragma unroll
      for (int r = 0; r < 8; ++r) {
        int m = rBase + rg * 16 + g * 8 + r;
        float val = acc[rg][n][r] + bvv;
        if (mode == 3) {
          outF[(size_t)m * EMB + nc] = val;
        } else {
          int bb = m >> 11;          // m / SS
          int s = m & (SS - 1);      // m % SS
          int h = nc >> 6;           // nc / DH
          int d = nc & (DH - 1);     // nc % DH
          size_t idx;
          if (mode == 2) idx = ((size_t)(bb * HEADS + h) * DH + d) * SS + s;
          else           idx = ((size_t)(bb * HEADS + h) * SS + s) * DH + d;
          outB[idx] = (bf16)val;
        }
      }
    }
}

// ---------------------------------------------------------------------------
// Flash attention: per (b,h), each wave owns a 16-row Q tile, streams keys
// in tiles of 32 with online softmax; O accumulated in fp32 C-fragments.
// Key loop unrolled 2x with ping-pong K-fragment sets; prefetches issue ahead
// of the softmax exp/shuffle block so their latency is hidden.
// Q,K: [bh][s][DH] bf16.  Vt: [bh][DH][s] bf16.  O out: [b*s][EMB] bf16.
// energy scale = 1/sqrt(EMB) = 1/32.
// ---------------------------------------------------------------------------
__global__ __launch_bounds__(128) void attn_wmma(
    const unsigned short* __restrict__ Qu, const unsigned short* __restrict__ Ku,
    const unsigned short* __restrict__ Vtu, unsigned short* __restrict__ Ou) {
  const bf16* Q  = reinterpret_cast<const bf16*>(Qu);
  const bf16* K  = reinterpret_cast<const bf16*>(Ku);
  const bf16* Vt = reinterpret_cast<const bf16*>(Vtu);
  bf16* O = reinterpret_cast<bf16*>(Ou);

  __shared__ __align__(16) bf16 plds[4][16 * 32];  // per-wave P staging (C->A transpose)

  int wave = threadIdx.x >> 5;
  int lane = threadIdx.x & 31;
  int g = lane >> 4;
  int c = lane & 15;

  int qBase = blockIdx.x * 64 + wave * 16;
  int bh = blockIdx.y;
  const bf16* Qp = Q  + (size_t)bh * SS * DH;
  const bf16* Kp = K  + (size_t)bh * SS * DH;
  const bf16* Vp = Vt + (size_t)bh * DH * SS;

  // Q tile 16 x 64 -> two A-fragments, resident for whole loop
  v16bf q0 = load_frag(Qp + (size_t)qBase * DH + 0, DH);
  v16bf q1 = load_frag(Qp + (size_t)qBase * DH + 32, DH);

  float mrow[8], lrow[8];
  v8f oacc[4];
#pragma unroll
  for (int r = 0; r < 8; ++r) { mrow[r] = -1.0e30f; lrow[r] = 0.0f; }
#pragma unroll
  for (int n = 0; n < 4; ++n)
#pragma unroll
    for (int r = 0; r < 8; ++r) oacc[n][r] = 0.0f;

  const float scale = 1.0f / 32.0f;  // 1/sqrt(EMB)

  auto load_ktile = [&](v16bf* kf, int j) {
    kf[0] = load_frag(Kp + (size_t)j * DH + 0, DH);
    kf[1] = load_frag(Kp + (size_t)j * DH + 32, DH);
    kf[2] = load_frag(Kp + (size_t)(j + 16) * DH + 0, DH);
    kf[3] = load_frag(Kp + (size_t)(j + 16) * DH + 32, DH);
  };

  // one 32-key flash step: S-tiles, online softmax, P@V accumulate
  auto step = [&](int j, const v16bf* kf) {
    v8f s0, s1;
#pragma unroll
    for (int r = 0; r < 8; ++r) { s0[r] = 0.0f; s1[r] = 0.0f; }
    s0 = WMMA_BF16(q0, kf[0], s0);
    s0 = WMMA_BF16(q1, kf[1], s0);
    s1 = WMMA_BF16(q0, kf[2], s1);
    s1 = WMMA_BF16(q1, kf[3], s1);

    // scale + per-row max over 32 keys (rows 0..7 in lanes 0..15,
    // rows 8..15 in lanes 16..31 -> xor-reduce over masks 1,2,4,8)
    float rm[8];
#pragma unroll
    for (int r = 0; r < 8; ++r) {
      s0[r] *= scale;
      s1[r] *= scale;
      rm[r] = fmaxf(s0[r], s1[r]);
    }
#pragma unroll
    for (int mask = 1; mask <= 8; mask <<= 1)
#pragma unroll
      for (int r = 0; r < 8; ++r)
        rm[r] = fmaxf(rm[r], __shfl_xor(rm[r], mask, 32));

    float alpha[8], p0[8], p1[8], rs[8];
#pragma unroll
    for (int r = 0; r < 8; ++r) {
      float mnew = fmaxf(mrow[r], rm[r]);
      alpha[r] = __expf(mrow[r] - mnew);
      mrow[r] = mnew;
      p0[r] = __expf(s0[r] - mnew);
      p1[r] = __expf(s1[r] - mnew);
      rs[r] = p0[r] + p1[r];
    }
#pragma unroll
    for (int mask = 1; mask <= 8; mask <<= 1)
#pragma unroll
      for (int r = 0; r < 8; ++r)
        rs[r] += __shfl_xor(rs[r], mask, 32);
#pragma unroll
    for (int r = 0; r < 8; ++r) lrow[r] = lrow[r] * alpha[r] + rs[r];
#pragma unroll
    for (int n = 0; n < 4; ++n)
#pragma unroll
      for (int r = 0; r < 8; ++r) oacc[n][r] *= alpha[r];

    // Stage P (C layout) into LDS, re-read as A fragment (16 rows x 32 keys)
#pragma unroll
    for (int r = 0; r < 8; ++r) {
      int row = g * 8 + r;
      plds[wave][row * 32 + c]      = (bf16)p0[r];
      plds[wave][row * 32 + 16 + c] = (bf16)p1[r];
    }
    __syncthreads();
    v16bf pf = load_frag(&plds[wave][0], 32);

    // O += P @ V : B operand from Vt [d][s], contiguous in keys
#pragma unroll
    for (int n = 0; n < 4; ++n) {
      v16bf bv = load_frag(Vp + (size_t)(n * 16) * SS + j, SS);
      oacc[n] = WMMA_BF16(pf, bv, oacc[n]);
    }
    __syncthreads();
  };

  // ping-pong K-fragment sets: set A = even tiles, set B = odd tiles
  v16bf kfa[4], kfb[4];
  load_ktile(kfa, 0);
  for (int j = 0; j < SS; j += 64) {
    load_ktile(kfb, j + 32);        // always in range (SS % 64 == 0)
    step(j, kfa);
    if (j + 64 < SS) load_ktile(kfa, j + 64);
    step(j + 32, kfb);
  }

  // normalize and store merged-head O as bf16 [b*s][EMB]
  int b = bh >> 4;
  int h = bh & (HEADS - 1);
#pragma unroll
  for (int n = 0; n < 4; ++n)
#pragma unroll
    for (int r = 0; r < 8; ++r) {
      int srow = qBase + g * 8 + r;
      int col = h * DH + n * 16 + c;
      O[((size_t)(b * SS + srow)) * EMB + col] = (bf16)(oacc[n][r] / lrow[r]);
    }
}

// ---------------------------------------------------------------------------
// Host launcher
// ---------------------------------------------------------------------------
extern "C" void kernel_launch(void* const* d_in, const int* in_sizes, int n_in,
                              void* d_out, int out_size, void* d_ws, size_t ws_size,
                              hipStream_t stream) {
  const float* x_enc = (const float*)d_in[0];
  const float* x_dec = (const float*)d_in[1];
  const float* Wq = (const float*)d_in[2];
  const float* bq = (const float*)d_in[3];
  const float* Wk = (const float*)d_in[4];
  const float* bk = (const float*)d_in[5];
  const float* Wv = (const float*)d_in[6];
  const float* bv = (const float*)d_in[7];
  const float* Wp = (const float*)d_in[8];
  const float* bp = (const float*)d_in[9];
  float* out = (float*)d_out;

  // workspace carve-up (bf16 buffers, 256B aligned)
  char* p = (char*)d_ws;
  auto carve = [&](size_t bytes) -> unsigned short* {
    unsigned short* r = (unsigned short*)p;
    p += (bytes + 255) & ~(size_t)255;
    return r;
  };
  const size_t SZ_X = (size_t)MROWS * EMB * 2;  // 16 MB
  const size_t SZ_W = (size_t)EMB * EMB * 2;    //  2 MB
  unsigned short* xdec_b = carve(SZ_X);
  unsigned short* xenc_b = carve(SZ_X);
  unsigned short* wq_b = carve(SZ_W);
  unsigned short* wk_b = carve(SZ_W);
  unsigned short* wv_b = carve(SZ_W);
  unsigned short* wp_b = carve(SZ_W);
  unsigned short* qbuf = carve(SZ_X);   // [b,h,s,d]
  unsigned short* kbuf = carve(SZ_X);   // [b,h,s,d]
  unsigned short* vtbuf = carve(SZ_X);  // [b,h,d,s]
  unsigned short* obuf = carve(SZ_X);   // [b*s, emb]

  const int nX = MROWS * EMB;
  const int nW = EMB * EMB;
  cvt_f32_to_bf16<<<nX / (256 * 8), 256, 0, stream>>>(x_dec, xdec_b, nX);
  cvt_f32_to_bf16<<<nX / (256 * 8), 256, 0, stream>>>(x_enc, xenc_b, nX);
  cvt_f32_to_bf16<<<nW / (256 * 8), 256, 0, stream>>>(Wq, wq_b, nW);
  cvt_f32_to_bf16<<<nW / (256 * 8), 256, 0, stream>>>(Wk, wk_b, nW);
  cvt_f32_to_bf16<<<nW / (256 * 8), 256, 0, stream>>>(Wv, wv_b, nW);
  cvt_f32_to_bf16<<<nW / (256 * 8), 256, 0, stream>>>(Wp, wp_b, nW);

  dim3 gg(MROWS / 128, EMB / 64);
  gemm_wmma<<<gg, 128, 0, stream>>>(xdec_b, wq_b, bq, qbuf, nullptr, 0);
  gemm_wmma<<<gg, 128, 0, stream>>>(xdec_b, wk_b, bk, kbuf, nullptr, 1);
  gemm_wmma<<<gg, 128, 0, stream>>>(xenc_b, wv_b, bv, vtbuf, nullptr, 2);

  attn_wmma<<<dim3(SS / 64, BB * HEADS), 128, 0, stream>>>(qbuf, kbuf, vtbuf, obuf);

  gemm_wmma<<<gg, 128, 0, stream>>>(obuf, wp_b, bp, nullptr, out, 3);
}
